// BiLSTM_CRF_S_87651692576921
// MI455X (gfx1250) — compile-verified, matrix operand order checked by reading, not run on hardware
//
#include <hip/hip_runtime.h>
#include <hip/hip_bf16.h>
#include <math.h>

// ---------------- CDNA5 WMMA types ----------------
typedef __attribute__((ext_vector_type(16))) __bf16 v16bf;
typedef __attribute__((ext_vector_type(8)))  __bf16 v8bf;
typedef __attribute__((ext_vector_type(8)))  float  v8f;
typedef __attribute__((ext_vector_type(4)))  int    v4i;

#define T_LEN 512
#define BATCH 128
#define DIM   256
#define HDIM  256          // per-direction hidden
#define GATES 1024         // 4*HDIM
#define LAB   32
#define START_TAG 30
#define NEG_INF  -10000.0f

// async LDS<->global path (guarded: falls back to ds_load+global_store)
#if defined(__AMDGCN__) && __has_builtin(__builtin_amdgcn_global_store_async_from_lds_b128) && \
    __has_builtin(__builtin_amdgcn_s_wait_asynccnt)
#define HAVE_ASYNC_LDS 1
using gptr4 = __attribute__((address_space(1))) v4i*;   // global int4*
using lptr4 = __attribute__((address_space(3))) v4i*;   // LDS int4*
#else
#define HAVE_ASYNC_LDS 0
#endif

__device__ __forceinline__ v8f wmma_bf16(v16bf a, v16bf b, v8f c) {
  // D = A(16x32) * B(32x16) + C, f32 accumulate
  return __builtin_amdgcn_wmma_f32_16x16x32_bf16(false, a, false, b, (short)0, c,
                                                 false, false);
}
__device__ __forceinline__ float sigmoidf_(float x) { return 1.0f / (1.0f + __expf(-x)); }

// ---------------- fp32 -> bf16 convert ----------------
__global__ void __launch_bounds__(256) k_f2bf(const float* __restrict__ src,
                                              __bf16* __restrict__ dst, int n) {
  for (int i = blockIdx.x * blockDim.x + threadIdx.x; i < n; i += gridDim.x * blockDim.x)
    dst[i] = (__bf16)src[i];
}

// ---------------- embed gather + input projection (both directions) --------
// xproj stored FRAGMENT-MAJOR: per t, 512 tiles (m=0..7, nt=0..63); each tile is
// [lane][8] bf16 = the exact 16x16 C/D accumulator layout -> single b128 per tile.
__global__ void __launch_bounds__(256) k_embproj(const __bf16* __restrict__ emb,
                                                 const __bf16* __restrict__ wih0,
                                                 const __bf16* __restrict__ wih1,
                                                 const float*  __restrict__ b0,
                                                 const float*  __restrict__ b1,
                                                 const int*    __restrict__ sent,
                                                 __bf16* __restrict__ xp0,
                                                 __bf16* __restrict__ xp1) {
  const int lane  = threadIdx.x & 31;
  const int mtile = blockIdx.x * 8 + (threadIdx.x >> 5);   // 4096 M-tiles over 65536 rows
  const int row0  = mtile * 16;
  const int half  = (lane >> 4) << 4;                      // K half: 0 or 16
  const int t     = mtile >> 3;                            // 8 M-tiles per timestep
  const int mloc  = mtile & 7;
  const int tok   = sent[row0 + (lane & 15)];
  const __bf16* abase = emb + (size_t)tok * DIM + half;
  v16bf afr[8];
  #pragma unroll
  for (int k = 0; k < 8; ++k) afr[k] = *(const v16bf*)(abase + k * 32);

  const __bf16* wih[2] = {wih0, wih1};
  const float*  bia[2] = {b0, b1};
  __bf16*       xp [2] = {xp0, xp1};

  for (int d = 0; d < 2; ++d) {
    for (int nt = 0; nt < 64; ++nt) {
      const int ncol = nt * 16 + (lane & 15);
      const float bv = bia[d][ncol];
      v8f acc;
      #pragma unroll
      for (int r = 0; r < 8; ++r) acc[r] = bv;
      const __bf16* bbase = wih[d] + (size_t)ncol * DIM + half;
      v16bf bcur = *(const v16bf*)(bbase);
      #pragma unroll
      for (int k = 0; k < 8; ++k) {                 // rolling k+1 prefetch
        v16bf bnxt = (k < 7) ? *(const v16bf*)(bbase + (k + 1) * 32) : bcur;
        acc = wmma_bf16(afr[k], bcur, acc);
        bcur = bnxt;
      }
      v8bf o;
      #pragma unroll
      for (int r = 0; r < 8; ++r) o[r] = (__bf16)acc[r];
      // fragment-major store: tile index = m*64 + nt, 256 bf16 per tile
      *(v8bf*)(xp[d] + ((size_t)t * 512 + mloc * 64 + nt) * 256 + lane * 8) = o;
    }
  }
}

// ---------------- persistent LSTM recurrence (1 block = 1 direction) --------
// 32 waves: wave w -> jt = w>>1 (hidden-unit tile), mg = w&1 (4 batch M-tiles).
// Each wave computes all 4 gate tiles for its (m, jt): c/h update in registers.
__global__ void __launch_bounds__(1024) k_lstm(const __bf16* __restrict__ xproj2,
                                               const __bf16* __restrict__ whh2,
                                               const float*  __restrict__ h0,
                                               const float*  __restrict__ c0,
                                               __bf16* __restrict__ hout2) {
  extern __shared__ __attribute__((aligned(32))) __bf16 hbuf[];  // [2][128][256]
  const int dir = blockIdx.x;
  const __bf16* xp  = xproj2 + (size_t)dir * T_LEN * BATCH * GATES;
  const __bf16* whh = whh2   + (size_t)dir * GATES * HDIM;
  const float*  h0d = h0 + (size_t)dir * BATCH * HDIM;
  const float*  c0d = c0 + (size_t)dir * BATCH * HDIM;
  __bf16* ho = hout2 + (size_t)dir * T_LEN * BATCH * HDIM;

  const int tid  = threadIdx.x;
  const int lane = tid & 31;
  const int w    = tid >> 5;
  const int jt   = w >> 1;                 // 0..15
  const int mg   = w & 1;                  // 0..1
  const int half = (lane >> 4) << 4;
  const int colj = jt * 16 + (lane & 15);  // hidden-unit column

  for (int i = tid; i < BATCH * HDIM; i += 1024) hbuf[i] = (__bf16)h0d[i];

  float cst[4][8];
  #pragma unroll
  for (int mi = 0; mi < 4; ++mi) {
    const int m = mg * 4 + mi;
    #pragma unroll
    for (int r = 0; r < 8; ++r)
      cst[mi][r] = c0d[(m * 16 + r + ((lane >> 4) << 3)) * HDIM + colj];
  }
  __syncthreads();

  for (int step = 0; step < T_LEN; ++step) {
    const int tt = dir ? (T_LEN - 1 - step) : step;
    const __bf16* hr = hbuf + (step & 1) * (BATCH * HDIM);
    __bf16*       hw = hbuf + ((step & 1) ^ 1) * (BATCH * HDIM);
    // hw was async-stored to global 2 steps ago: wait before overwriting it
#if HAVE_ASYNC_LDS
    __builtin_amdgcn_s_wait_asynccnt(0);
#endif
    const __bf16* xpt = xp + (size_t)tt * 512 * 256;   // fragment-major slab
    if (step + 1 < T_LEN) {    // prefetch next timestep's gate projections
      const int tn = dir ? (tt - 1) : (tt + 1);
      __builtin_prefetch(xp + (size_t)tn * 512 * 256 + (size_t)tid * 128, 0, 1);
    }
    #pragma unroll
    for (int mi = 0; mi < 4; ++mi) {
      const int m = mg * 4 + mi;
      const __bf16* ab = hr + (m * 16 + (lane & 15)) * HDIM + half;
      v16bf afr[8];
      #pragma unroll
      for (int k = 0; k < 8; ++k) afr[k] = *(const v16bf*)(ab + k * 32);
      v8f gacc[4];
      #pragma unroll
      for (int g = 0; g < 4; ++g) {
        const int ncol = g * 256 + colj;
        // C seed: one b128 fragment load (tile = m*64 + g*16 + jt)
        v8bf cf = *(const v8bf*)(xpt + (size_t)(m * 64 + g * 16 + jt) * 256 + lane * 8);
        v8f acc;
        #pragma unroll
        for (int r = 0; r < 8; ++r) acc[r] = (float)cf[r];
        const __bf16* bb = whh + (size_t)ncol * HDIM + half;
        v16bf bcur = *(const v16bf*)(bb);
        #pragma unroll
        for (int k = 0; k < 8; ++k) {               // rolling k+1 prefetch
          v16bf bnxt = (k < 7) ? *(const v16bf*)(bb + (k + 1) * 32) : bcur;
          acc = wmma_bf16(afr[k], bcur, acc);
          bcur = bnxt;
        }
        gacc[g] = acc;
      }
      #pragma unroll
      for (int r = 0; r < 8; ++r) {                 // torch gate order i,f,g,o
        const float ig = sigmoidf_(gacc[0][r]);
        const float fg = sigmoidf_(gacc[1][r]);
        const float gg = tanhf(gacc[2][r]);
        const float og = sigmoidf_(gacc[3][r]);
        const float c  = fg * cst[mi][r] + ig * gg;
        cst[mi][r] = c;
        const float h = og * tanhf(c);
        const int row = m * 16 + r + ((lane >> 4) << 3);
        hw[row * HDIM + colj] = (__bf16)h;
      }
    }
    __syncthreads();   // hw complete; hr reads complete
    // cooperative 64KB h writeback: 1024 threads x 32 bf16 (4 x b128)
    {
      __bf16* hog = ho + (size_t)tt * BATCH * HDIM + tid * 32;
      const __bf16* src = hw + tid * 32;
      #pragma unroll
      for (int q = 0; q < 4; ++q) {
#if HAVE_ASYNC_LDS
        __builtin_amdgcn_global_store_async_from_lds_b128(
            (gptr4)(hog + q * 8), (lptr4)(src + q * 8), 0, 0);
#else
        *(v8bf*)(hog + q * 8) = *(const v8bf*)(src + q * 8);
#endif
      }
    }
  }
  // S_ENDPGM implies wait-idle; outstanding async stores drain before kernel end.
}

// ---------------- output projection: feats = [hf|hb] @ Wo^T + bo ------------
__global__ void __launch_bounds__(256) k_feats(const __bf16* __restrict__ hf,
                                               const __bf16* __restrict__ hb,
                                               const __bf16* __restrict__ wo,
                                               const float*  __restrict__ bo,
                                               float* __restrict__ feats) {
  const int lane  = threadIdx.x & 31;
  const int mtile = blockIdx.x * 8 + (threadIdx.x >> 5);
  const int row0  = mtile * 16;
  const int half  = (lane >> 4) << 4;
  const int arow  = row0 + (lane & 15);
  v16bf afr[16];
  #pragma unroll
  for (int k = 0; k < 8; ++k) afr[k]     = *(const v16bf*)(hf + (size_t)arow * HDIM + half + k * 32);
  #pragma unroll
  for (int k = 0; k < 8; ++k) afr[8 + k] = *(const v16bf*)(hb + (size_t)arow * HDIM + half + k * 32);
  #pragma unroll
  for (int nt = 0; nt < 2; ++nt) {
    const int ncol = nt * 16 + (lane & 15);
    const float bv = bo[ncol];
    v8f acc;
    #pragma unroll
    for (int r = 0; r < 8; ++r) acc[r] = bv;
    const __bf16* bbase = wo + (size_t)ncol * 512 + half;
    v16bf bcur = *(const v16bf*)(bbase);
    #pragma unroll
    for (int k = 0; k < 16; ++k) {
      v16bf bnxt = (k < 15) ? *(const v16bf*)(bbase + (k + 1) * 32) : bcur;
      acc = wmma_bf16(afr[k], bcur, acc);
      bcur = bnxt;
    }
    #pragma unroll
    for (int r = 0; r < 8; ++r)
      feats[(size_t)(row0 + r + ((lane >> 4) << 3)) * LAB + ncol] = acc[r];
  }
}

// ---------------- CRF forward + gold score + final scalar -------------------
__global__ void __launch_bounds__(1024) k_crf(const float* __restrict__ feats,
                                              const float* __restrict__ trans,
                                              const int*   __restrict__ labels,
                                              const unsigned char* __restrict__ masks,
                                              float* __restrict__ out) {
  __shared__ float dp[2][BATCH * LAB];
  __shared__ float ts[LAB * LAB];
  __shared__ float gold[BATCH];
  __shared__ int   last[BATCH];
  __shared__ float red[BATCH];
  const int tid = threadIdx.x;
  if (tid < LAB * LAB) ts[tid] = trans[tid];
  for (int i = tid; i < BATCH * LAB; i += 1024)
    dp[0][i] = ((i & 31) == START_TAG) ? 0.0f : NEG_INF;
  if (tid < BATCH) { gold[tid] = 0.0f; last[tid] = START_TAG; }
  __syncthreads();

  for (int t = 0; t < T_LEN; ++t) {
    const float* sc = feats + (size_t)t * BATCH * LAB;
    const int cur = t & 1, nxt = cur ^ 1;
    #pragma unroll
    for (int i = 0; i < 4; ++i) {
      const int p = tid + i * 1024;
      const int b = p >> 5, n = p & 31;
      float v;
      if (masks[t * BATCH + b]) {
        float mx = -3.4e38f;
        #pragma unroll
        for (int m = 0; m < LAB; ++m)
          mx = fmaxf(mx, dp[cur][b * LAB + m] + ts[m * LAB + n]);
        float s = 0.0f;
        #pragma unroll
        for (int m = 0; m < LAB; ++m)
          s += __expf(dp[cur][b * LAB + m] + ts[m * LAB + n] - mx);
        v = sc[b * LAB + n] + mx + __logf(s);
      } else {
        v = dp[cur][b * LAB + n];
      }
      dp[nxt][b * LAB + n] = v;
    }
    if (tid < BATCH) {
      const int b = tid;
      if (masks[t * BATCH + b]) {
        const int ln = labels[t * BATCH + b];
        gold[b] += ts[last[b] * LAB + ln] + sc[b * LAB + ln];
        last[b] = ln;
      }
    }
    __syncthreads();
  }
  if (tid < BATCH) {
    const int b = tid;
    const float* dpf = dp[T_LEN & 1];
    float mx = -3.4e38f;
    for (int m = 0; m < LAB; ++m) mx = fmaxf(mx, dpf[b * LAB + m]);
    float s = 0.0f;
    for (int m = 0; m < LAB; ++m) s += __expf(dpf[b * LAB + m] - mx);
    red[b] = (mx + __logf(s)) - gold[b];
  }
  __syncthreads();
  if (tid == 0) {
    float s = 0.0f;
    for (int b = 0; b < BATCH; ++b) s += red[b];
    out[0] = s / (float)BATCH;
  }
}

// ---------------- host-side orchestration ----------------------------------
extern "C" void kernel_launch(void* const* d_in, const int* in_sizes, int n_in,
                              void* d_out, int out_size, void* d_ws, size_t ws_size,
                              hipStream_t stream) {
  (void)in_sizes; (void)n_in; (void)out_size; (void)ws_size;
  const float* emb    = (const float*)d_in[0];
  const float* wihf   = (const float*)d_in[1];
  const float* whhf   = (const float*)d_in[2];
  const float* bf     = (const float*)d_in[3];
  const float* wihb   = (const float*)d_in[4];
  const float* whhb   = (const float*)d_in[5];
  const float* bb     = (const float*)d_in[6];
  const float* wo     = (const float*)d_in[7];
  const float* bo     = (const float*)d_in[8];
  const float* trans  = (const float*)d_in[9];
  const float* h0     = (const float*)d_in[10];
  const float* c0     = (const float*)d_in[11];
  const int*   sent   = (const int*)d_in[12];
  const int*   labels = (const int*)d_in[13];
  const unsigned char* masks = (const unsigned char*)d_in[14];

  char* ws = (char*)d_ws;
  __bf16* emb_bf = (__bf16*)(ws);                                   //  8,192,000 B
  __bf16* wih_bf = (__bf16*)(ws + 8192000);                         //  2 x 524,288 B
  __bf16* whh_bf = (__bf16*)(ws + 8192000 + 1048576);               //  2 x 524,288 B
  __bf16* wo_bf  = (__bf16*)(ws + 8192000 + 2097152);               //     32,768 B
  __bf16* xproj  = (__bf16*)(ws + 10321920);                        // 268,435,456 B
  __bf16* hout   = (__bf16*)(ws + 10321920 + 268435456);            //  67,108,864 B
  float*  feats  = (float*) (ws + 10321920 + 268435456 + 67108864); //   8,388,608 B

  // fp32 -> bf16 operand staging (weights stay L2-resident)
  k_f2bf<<<dim3(1024), dim3(256), 0, stream>>>(emb,  emb_bf,          16000 * 256);
  k_f2bf<<<dim3(256),  dim3(256), 0, stream>>>(wihf, wih_bf,          1024 * 256);
  k_f2bf<<<dim3(256),  dim3(256), 0, stream>>>(wihb, wih_bf + 262144, 1024 * 256);
  k_f2bf<<<dim3(256),  dim3(256), 0, stream>>>(whhf, whh_bf,          1024 * 256);
  k_f2bf<<<dim3(256),  dim3(256), 0, stream>>>(whhb, whh_bf + 262144, 1024 * 256);
  k_f2bf<<<dim3(16),   dim3(256), 0, stream>>>(wo,   wo_bf,           32 * 512);

  // fused input projections: one gather, both directions (WMMA GEMM)
  k_embproj<<<dim3(512), dim3(256), 0, stream>>>(
      emb_bf, wih_bf, wih_bf + 262144, bf, bb, sent,
      xproj, xproj + (size_t)T_LEN * BATCH * GATES);

  // sequential recurrences: one WGP-resident block per direction, 128KB LDS
  k_lstm<<<dim3(2), dim3(1024), 2 * BATCH * HDIM * (int)sizeof(__bf16), stream>>>(
      xproj, whh_bf, h0, c0, hout);

  // output projection (WMMA GEMM)
  k_feats<<<dim3(512), dim3(256), 0, stream>>>(hout, hout + (size_t)T_LEN * BATCH * HDIM,
                                               wo_bf, bo, feats);

  // CRF scan + loss scalar
  k_crf<<<dim3(1), dim3(1024), 0, stream>>>(feats, trans, labels, masks, (float*)d_out);
}
